// MDConv_59227599012133
// MI455X (gfx1250) — compile-verified, matrix operand order checked by reading, not run on hardware
//
#include <hip/hip_runtime.h>
#include <hip/hip_bf16.h>
#include <cstdint>

typedef __attribute__((ext_vector_type(16))) __bf16       v16bf;
typedef __attribute__((ext_vector_type(8)))  float        v8f;
typedef __attribute__((ext_vector_type(4)))  float        v4f;
typedef __attribute__((ext_vector_type(4)))  unsigned int v4u;

#define B_    8
#define CIN   256
#define COUT  256
#define H_    64
#define W_    64
#define KTAPS 9
#define KDIM  (CIN * KTAPS)     // 2304, K order PERMUTED: g = tap*256 + i
#define NPIX  (H_ * W_)         // 4096
#define KSTEP 32
#define NKST  (KDIM / KSTEP)    // 72  (9 taps x 8 channel-blocks)

// Native bf16 convert: lets clang pick v_cvt_*bf16* ops on gfx1250
// (adjacent pairs in Pack8 can fuse into packed converts).
__device__ __forceinline__ unsigned short f32_bf16(float f) {
    return __builtin_bit_cast(unsigned short, (__bf16)f);
}

union Frag16 { v16bf v; v4u q[2]; };           // 32 bytes: bf16 A/B fragment
union Pack8  { unsigned short u[8]; v4u q; };  // 16-byte LDS store helper

// ---------------------------------------------------------------------------
// Pack weights to bf16 with K permuted to tap-major: dst[o][tap*256+i].
// w layout in memory: [o][i][tap] (OIHW flattened), so src = o*2304 + i*9 + tap.
// ---------------------------------------------------------------------------
__global__ void k_pack(const float* __restrict__ w_dcn, const float* __restrict__ w_off,
                       unsigned short* __restrict__ pk_dcn, unsigned short* __restrict__ pk_off) {
    const int ND = COUT * KDIM;       // 589824
    const int NO = 32 * KDIM;         // 73728 (rows 27..31 zero-padded)
    int idx = blockIdx.x * blockDim.x + threadIdx.x;
    if (idx < ND) {
        int o   = idx / KDIM;
        int r   = idx - o * KDIM;
        int tap = r >> 8;             // r / 256
        int i   = r & 255;
        pk_dcn[idx] = f32_bf16(w_dcn[o * KDIM + i * KTAPS + tap]);
    } else if (idx < ND + NO) {
        int j   = idx - ND;
        int c   = j / KDIM;
        int r   = j - c * KDIM;
        int tap = r >> 8;
        int i   = r & 255;
        pk_off[j] = (c < 27) ? f32_bf16(w_off[c * KDIM + i * KTAPS + tap])
                             : (unsigned short)0;
    }
}

// ---------------------------------------------------------------------------
// Offset conv as WMMA GEMM: M=32 (27 live rows), K=2304 (tap-major), N=64
// pixels (one image row) per WG. 8 waves in 2(M) x 4(N), one 16x16 acc each.
// Branchless im2col: clamped address, multiplicative validity mask.
// ---------------------------------------------------------------------------
__global__ __launch_bounds__(256) void k_offconv(const float* __restrict__ x,
        const unsigned short* __restrict__ pk_off, const float* __restrict__ b_off,
        float* __restrict__ off) {
    __shared__ unsigned short Bl[2][64 * 40];  // double-buffered, 80B rows
    __shared__ float bsh[32];                  // staged bias

    const int tid  = threadIdx.x;
    const int blk  = blockIdx.x;
    const int b    = blk >> 6;
    const int oy   = blk & 63;
    const int lane = tid & 31, wave = tid >> 5;
    const int mt   = wave >> 2, nt = wave & 3;
    const int khalf = lane >> 4, lan16 = lane & 15;
    const int n_b  = tid & 63;                 // B-build: pixel column
    const int kk0  = (tid >> 6) << 3;          // B-build: 8 consecutive kk

    if (tid < 32) bsh[tid] = (tid < 27) ? b_off[tid] : 0.f;

    const float* xb = x + (size_t)b * CIN * NPIX;
    const unsigned short* arow = pk_off + (mt * 16 + lan16) * KDIM;
    const unsigned short* brow0 = &Bl[0][0] + (nt * 16 + lan16) * 40 + khalf * 8;

    v8f acc = {};
    for (int ks = 0; ks < NKST; ++ks) {
        const int tap = ks >> 3;               // uniform within the step
        const int i0  = (ks & 7) << 5;         // base input channel
        const int ky  = tap / 3, kx = tap - ky * 3;
        const int iy  = oy + ky - 1;
        const int ix  = n_b + kx - 1;
        const bool ok = ((unsigned)iy < 64u) && ((unsigned)ix < 64u);
        const float vm = ok ? 1.f : 0.f;
        const int iyc = iy < 0 ? 0 : (iy > 63 ? 63 : iy);
        const int ixc = ix < 0 ? 0 : (ix > 63 ? 63 : ix);
        const float* xs = xb + ((size_t)(i0 + kk0) << 12) + (iyc << 6) + ixc;

        Pack8 t8;
        #pragma unroll
        for (int j = 0; j < 8; ++j)            // 8 unconditional strided loads
            t8.u[j] = f32_bf16(vm * fmaxf(0.f, xs[(size_t)j << 12]));
        *(v4u*)(&Bl[ks & 1][0] + n_b * 40 + kk0) = t8.q;
        __syncthreads();                       // single barrier (double buffer)

        int kb = ks * KSTEP + khalf * 8;
        Frag16 af, bfr;
        af.q[0] = *(const v4u*)(arow + kb);
        af.q[1] = *(const v4u*)(arow + kb + 16);
        const unsigned short* br = brow0 + (ks & 1) * (64 * 40);
        bfr.q[0] = *(const v4u*)(br);
        bfr.q[1] = *(const v4u*)(br + 16);
        acc = __builtin_amdgcn_wmma_f32_16x16x32_bf16(false, af.v, false, bfr.v,
                                                      (short)0, acc, false, false);
    }

    // ---- epilogue: bias from LDS, predicated store only ----
    const int gn  = nt * 16 + lan16;
    const int cb  = mt * 16 + ((lane >> 4) << 3);   // base channel for this lane
    float* ob = off + (((size_t)b * 27 + cb) << 12) + (oy << 6) + gn;
    #pragma unroll
    for (int r = 0; r < 8; ++r) {
        int   c = cb + r;
        float v = acc[r] + bsh[c];
        if (c < 27) ob[(size_t)r << 12] = v;
    }
}

// ---------------------------------------------------------------------------
// Fused bilinear-sample + main GEMM: M=256, K=2304 (tap-major), N=64 pixels.
// 8 waves in 4(M) x 2(N); each wave 4x2 16x16 f32 accumulators.
// Per K-step the tap is uniform -> one 16B idx + one 16B weight LDS load per
// thread, then 32 unconditional gathers (weights already encode validity+mask).
// ---------------------------------------------------------------------------
__global__ __launch_bounds__(256) void k_dcn(const float* __restrict__ x,
        const float* __restrict__ off, const unsigned short* __restrict__ pk_dcn,
        float* __restrict__ y) {
    __shared__ unsigned short Bl[2][64 * 40];  // 10 KB double-buffered B tile
    __shared__ int   sidx[64 * 9 * 4];         //  9 KB bilinear corner indices
    __shared__ float swt [64 * 9 * 4];         //  9 KB bilinear weights * mask

    const int tid = threadIdx.x;
    const int blk = blockIdx.x;
    const int b   = blk >> 6;
    const int oy  = blk & 63;

    // ---- phase 1: per-(pixel,tap) sample table (runs once) ----
    const float* offb = off + (((size_t)b * 27) << 12) + (oy << 6);
    for (int e = tid; e < 64 * KTAPS; e += 256) {
        int n = e / 9, k = e - n * 9;
        float dy  = offb[((2 * k    ) << 12) + n];
        float dx  = offb[((2 * k + 1) << 12) + n];
        float mm  = offb[((18 + k   ) << 12) + n];
        float msk = 1.f / (1.f + __expf(-mm));
        float ysv = dy + (float)(k / 3) + (float)oy - 1.f;
        float xsv = dx + (float)(k % 3) + (float)n  - 1.f;
        float y0 = floorf(ysv), x0 = floorf(xsv);
        float fy = ysv - y0,    fx = xsv - x0;
        int iy0 = (int)y0, ix0 = (int)x0;
        #pragma unroll
        for (int c = 0; c < 4; ++c) {
            int   yc = iy0 + (c >> 1);
            int   xc = ix0 + (c & 1);
            float wb = ((c >> 1) ? fy : 1.f - fy) * ((c & 1) ? fx : 1.f - fx);
            bool  okc = ((unsigned)yc < 64u) && ((unsigned)xc < 64u);
            sidx[e * 4 + c] = okc ? ((yc << 6) + xc) : 0;
            swt [e * 4 + c] = okc ? wb * msk : 0.f;
        }
    }
    __syncthreads();

    const int lane = tid & 31, wave = tid >> 5;
    const int mq = wave & 3;                   // 64-row M block
    const int nh = wave >> 2;                  // 32-col N half
    const int khalf = lane >> 4, lan16 = lane & 15;
    const int n_b = tid & 63, kk0 = (tid >> 6) << 3;
    const float* xb = x + (size_t)b * CIN * NPIX;

    v8f acc[4][2];
    #pragma unroll
    for (int mt = 0; mt < 4; ++mt)
        #pragma unroll
        for (int nt = 0; nt < 2; ++nt) acc[mt][nt] = (v8f){};

    // ---- phase 2: K loop ----
    for (int ks = 0; ks < NKST; ++ks) {
        const int tap = ks >> 3;
        const int i0  = (ks & 7) << 5;
        const int e4  = (n_b * 9 + tap) * 4;
        const v4u id = *(const v4u*)(sidx + e4);   // 16B ds_load
        const v4f wt = *(const v4f*)(swt  + e4);   // 16B ds_load
        const float* xc = xb + ((size_t)(i0 + kk0) << 12);

        Pack8 t8;
        #pragma unroll
        for (int j = 0; j < 8; ++j) {              // 32 unconditional gathers
            const float* xj = xc + ((size_t)j << 12);
            float v = wt[0] * fmaxf(0.f, xj[id[0]])
                    + wt[1] * fmaxf(0.f, xj[id[1]])
                    + wt[2] * fmaxf(0.f, xj[id[2]])
                    + wt[3] * fmaxf(0.f, xj[id[3]]);
            t8.u[j] = f32_bf16(v);
        }
        *(v4u*)(&Bl[ks & 1][0] + n_b * 40 + kk0) = t8.q;
        __syncthreads();                           // single barrier per step

        Frag16 bfr[2];
        #pragma unroll
        for (int nt = 0; nt < 2; ++nt) {
            const unsigned short* br =
                &Bl[ks & 1][0] + (nh * 32 + nt * 16 + lan16) * 40 + khalf * 8;
            bfr[nt].q[0] = *(const v4u*)(br);
            bfr[nt].q[1] = *(const v4u*)(br + 16);
        }
        int kb = ks * KSTEP + khalf * 8;
        #pragma unroll
        for (int mt = 0; mt < 4; ++mt) {
            Frag16 af;
            const unsigned short* ar =
                pk_dcn + (size_t)(mq * 64 + mt * 16 + lan16) * KDIM + kb;
            af.q[0] = *(const v4u*)(ar);
            af.q[1] = *(const v4u*)(ar + 16);
            #pragma unroll
            for (int nt = 0; nt < 2; ++nt)
                acc[mt][nt] = __builtin_amdgcn_wmma_f32_16x16x32_bf16(
                    false, af.v, false, bfr[nt].v, (short)0, acc[mt][nt], false, false);
        }
    }

    // ---- phase 3: store y tile ----
    float* yb = y + (((size_t)b * COUT) << 12) + (oy << 6);
    #pragma unroll
    for (int mt = 0; mt < 4; ++mt)
        #pragma unroll
        for (int nt = 0; nt < 2; ++nt) {
            int gn = nh * 32 + nt * 16 + lan16;
            int gm = mq * 64 + mt * 16 + ((lane >> 4) << 3);
            float* yo = yb + ((size_t)gm << 12) + gn;
            #pragma unroll
            for (int r = 0; r < 8; ++r)
                yo[(size_t)r << 12] = acc[mt][nt][r];
        }
}

// ---------------------------------------------------------------------------
// BatchNorm: per-channel sum / sumsq (deterministic tree reduce), then apply.
// ---------------------------------------------------------------------------
__global__ __launch_bounds__(256) void k_bnstats(const float* __restrict__ y,
                                                 float* __restrict__ stats) {
    __shared__ float ss[256], s2[256];
    int c = blockIdx.x, tid = threadIdx.x;
    float a = 0.f, q = 0.f;
    for (int lin = tid; lin < B_ * NPIX; lin += 256) {
        int bb = lin >> 12, p = lin & (NPIX - 1);
        float v = y[(((size_t)bb * COUT + c) << 12) + p];
        a += v; q += v * v;
    }
    ss[tid] = a; s2[tid] = q;
    __syncthreads();
    for (int s = 128; s > 0; s >>= 1) {
        if (tid < s) { ss[tid] += ss[tid + s]; s2[tid] += s2[tid + s]; }
        __syncthreads();
    }
    if (tid == 0) { stats[c] = ss[0]; stats[256 + c] = s2[0]; }
}

__global__ void k_bnapply(const float* __restrict__ y, const float* __restrict__ stats,
                          const float* __restrict__ gamma, const float* __restrict__ beta,
                          float* __restrict__ out, int total) {
    const float inv = 1.f / (float)(B_ * NPIX);
    for (int idx = blockIdx.x * blockDim.x + threadIdx.x; idx < total;
         idx += gridDim.x * blockDim.x) {
        int c = (idx >> 12) & (COUT - 1);
        float mean = stats[c] * inv;
        float var  = stats[256 + c] * inv - mean * mean;
        out[idx] = (y[idx] - mean) * rsqrtf(var + 1e-5f) * gamma[c] + beta[c];
    }
}

// ---------------------------------------------------------------------------
extern "C" void kernel_launch(void* const* d_in, const int* in_sizes, int n_in,
                              void* d_out, int out_size, void* d_ws, size_t ws_size,
                              hipStream_t stream) {
    const float* x     = (const float*)d_in[0];
    const float* w_off = (const float*)d_in[1];
    const float* b_off = (const float*)d_in[2];
    const float* w_dcn = (const float*)d_in[3];
    const float* gamma = (const float*)d_in[4];
    const float* beta  = (const float*)d_in[5];

    char* ws = (char*)d_ws;
    unsigned short* pk_off = (unsigned short*)(ws + 0);          //   147456 B
    unsigned short* pk_dcn = (unsigned short*)(ws + 147456);     //  1179648 B
    float* off   = (float*)(ws + 1327104);                       //  3538944 B
    float* ybuf  = (float*)(ws + 4866048);                       // 33554432 B
    float* stats = (float*)(ws + 38420480);                      //     2048 B

    int packN = COUT * KDIM + 32 * KDIM;
    k_pack   <<<(packN + 255) / 256, 256, 0, stream>>>(w_dcn, w_off, pk_dcn, pk_off);
    k_offconv<<<B_ * H_, 256, 0, stream>>>(x, pk_off, b_off, off);
    k_dcn    <<<B_ * H_, 256, 0, stream>>>(x, off, pk_dcn, ybuf);
    k_bnstats<<<COUT, 256, 0, stream>>>(ybuf, stats);
    k_bnapply<<<2048, 256, 0, stream>>>(ybuf, stats, gamma, beta, (float*)d_out, out_size);
}
